// ImageFratureAlign_46127948759316
// MI455X (gfx1250) — matrix operand order repaired and checked by reading, not compiled, not run on hardware
//
#include <hip/hip_runtime.h>
#include <hip/hip_bf16.h>
#include <math.h>
#include <stdint.h>

typedef __bf16 bf16_t;
typedef __attribute__((ext_vector_type(16))) __bf16 v16bf;
typedef __attribute__((ext_vector_type(8)))  __bf16 v8bf;
typedef __attribute__((ext_vector_type(8)))  float  v8f;

// gfx1250 async global->LDS copy (ASYNCcnt-tracked), one 16B chunk per lane.
#define ASYNC_LOAD_B128(ldsoff, gaddr)                                         \
    asm volatile("global_load_async_to_lds_b128 %0, %1, off"                   \
                 :: "v"(ldsoff), "v"(gaddr) : "memory")

// ---------------------------------------------------------------------------
// Fragment builders following CDNA5 16-bit WMMA VGPR layouts (CDNA5 ISA)
// A (16x32, MxK): lane<16 -> row=lane, K in {0..7} U {16..23}; lane>=16 -> row=lane-16,
//                 K in {8..15} U {24..31}. Two 16B loads.
// B (32x16, KxN): lane holds column N=lane&15; same K set as A. Two 16B loads.
// ---------------------------------------------------------------------------
__device__ __forceinline__ v16bf frag_a(const bf16_t* p) {
    v8bf lo = *(const v8bf*)(p);
    v8bf hi = *(const v8bf*)(p + 16);
    return __builtin_shufflevector(lo, hi, 0,1,2,3,4,5,6,7,8,9,10,11,12,13,14,15);
}
__device__ __forceinline__ v16bf frag_b(const bf16_t* p) {
    v8bf lo = *(const v8bf*)(p);
    v8bf hi = *(const v8bf*)(p + 16);
    return __builtin_shufflevector(lo, hi, 0,1,2,3,4,5,6,7,8,9,10,11,12,13,14,15);
}

// ---------------------------------------------------------------------------
// GEMM: C[M,N] = relu(A[M,K](bf16) * Bw[N,K]^T(bf16) + bias[N])  (f32 out)
// Block = 256 threads (8 waves) -> 128x64 output tile; wave w owns rows w*16..w*16+15.
// K staged through LDS in 64-deep double-buffered stages via async-to-LDS copies:
// per wave per stage: 4 A-chunks + 2 B-chunks (ASYNCcnt), then 2x(1 A frag, 4 B frags,
// 4 wmma) from LDS. Rows padded to 72 bf16 (144B stride) for conflict-free ds reads.
// grid.x = 49 * (Ncols/64);  M fixed at 6272 rows.
// ---------------------------------------------------------------------------
__global__ void __launch_bounds__(256)
gemm_bias_relu_bf16(const bf16_t* __restrict__ A, const bf16_t* __restrict__ Bw,
                    const float* __restrict__ bias, float* __restrict__ C,
                    int Ncols, int K) {
    __shared__ __align__(16) bf16_t As[2][128][72];
    __shared__ __align__(16) bf16_t Bs[2][64][72];

    const int tid  = threadIdx.x;
    const int w    = tid >> 5;
    const int lane = tid & 31;
    const int bm   = blockIdx.x % 49;
    const int bn   = blockIdx.x / 49;
    const int m0   = bm << 7;
    const int n0   = bn << 6;

    // async-copy mapping: A: 2 threads/row, 64B (4 chunks) each; B: 4 threads/col, 32B each
    const int ar = tid >> 1, aseg = (tid & 1) << 5;      // element offsets within 64-K stage
    const int bc = tid >> 2, bseg = (tid & 3) << 4;
    const bf16_t* gA = A  + (size_t)(m0 + ar) * K + aseg;
    const bf16_t* gB = Bw + (size_t)(n0 + bc) * K + bseg;

    auto issue = [&](int buf, int k0) {
        const bf16_t* pa = gA + k0;
        uint32_t la = (uint32_t)(uintptr_t)&As[buf][ar][aseg];
        ASYNC_LOAD_B128(la,      (uint64_t)(uintptr_t)(pa));
        ASYNC_LOAD_B128(la + 16, (uint64_t)(uintptr_t)(pa + 8));
        ASYNC_LOAD_B128(la + 32, (uint64_t)(uintptr_t)(pa + 16));
        ASYNC_LOAD_B128(la + 48, (uint64_t)(uintptr_t)(pa + 24));
        const bf16_t* pb = gB + k0;
        uint32_t lb = (uint32_t)(uintptr_t)&Bs[buf][bc][bseg];
        ASYNC_LOAD_B128(lb,      (uint64_t)(uintptr_t)(pb));
        ASYNC_LOAD_B128(lb + 16, (uint64_t)(uintptr_t)(pb + 8));
    };

    // wmma lane mapping
    const int lrow = (w << 4) + (lane & 15);
    const int akb  = (lane >> 4) << 3;       // 0 or 8
    const int bcl  = lane & 15;
    const int bkb  = (lane >> 4) << 3;       // 0 or 8

    v8f acc0 = {}, acc1 = {}, acc2 = {}, acc3 = {};
    const int nst = K >> 6;                  // 64-deep stages (>= 2 for K in {1024,2048})
    issue(0, 0);
    issue(1, 64);
    for (int s = 0; s < nst; ++s) {
        const int buf = s & 1;
        if (s + 1 < nst) asm volatile("s_wait_asynccnt 6" ::: "memory");
        else             asm volatile("s_wait_asynccnt 0" ::: "memory");
        __syncthreads();
#pragma unroll
        for (int kk = 0; kk < 64; kk += 32) {
            v16bf a  = frag_a(&As[buf][lrow][kk + akb]);
            v16bf b0 = frag_b(&Bs[buf][bcl +  0][kk + bkb]);
            v16bf b1 = frag_b(&Bs[buf][bcl + 16][kk + bkb]);
            v16bf b2 = frag_b(&Bs[buf][bcl + 32][kk + bkb]);
            v16bf b3 = frag_b(&Bs[buf][bcl + 48][kk + bkb]);
            acc0 = __builtin_amdgcn_wmma_f32_16x16x32_bf16(false, a, false, b0, (short)0, acc0, false, false);
            acc1 = __builtin_amdgcn_wmma_f32_16x16x32_bf16(false, a, false, b1, (short)0, acc1, false, false);
            acc2 = __builtin_amdgcn_wmma_f32_16x16x32_bf16(false, a, false, b2, (short)0, acc2, false, false);
            acc3 = __builtin_amdgcn_wmma_f32_16x16x32_bf16(false, a, false, b3, (short)0, acc3, false, false);
        }
        __syncthreads();
        if (s + 2 < nst) issue(buf, (s + 2) << 6);
    }

    // C/D layout: VGPR r, lanes<16 -> M=r, lanes>=16 -> M=r+8; N = lane&15.
    v8f accs[4] = {acc0, acc1, acc2, acc3};
    const int rbase = m0 + (w << 4) + ((lane >> 4) << 3);
    const int ccol  = lane & 15;
#pragma unroll
    for (int j = 0; j < 4; ++j) {
        const int col = n0 + (j << 4) + ccol;
        const float bb = bias[col];
        float* cp = C + (size_t)rbase * Ncols + col;
#pragma unroll
        for (int r = 0; r < 8; ++r) {
            float v = accs[j][r] + bb;
            cp[(size_t)r * Ncols] = v > 0.f ? v : 0.f;
        }
    }
}

// ---------------------------------------------------------------------------
// BN stats: per n in [0,49): mean/var of H over (b in 0..127, c in 0..Ncols).
// ---------------------------------------------------------------------------
__global__ void __launch_bounds__(256)
bn_reduce(const float* __restrict__ H, float* __restrict__ stats, int Ncols) {
    __shared__ float ss[256], sq[256];
    const int n = blockIdx.x;
    const int tid = threadIdx.x;
    float s = 0.f, q = 0.f;
    for (int b = 0; b < 128; ++b) {
        const float* hp = H + (size_t)(b * 49 + n) * Ncols;
        for (int c = tid; c < Ncols; c += 256) {
            float v = hp[c];
            s += v; q += v * v;
        }
    }
    ss[tid] = s; sq[tid] = q; __syncthreads();
    for (int st = 128; st > 0; st >>= 1) {
        if (tid < st) { ss[tid] += ss[tid + st]; sq[tid] += sq[tid + st]; }
        __syncthreads();
    }
    if (tid == 0) {
        float cnt  = 128.f * (float)Ncols;
        float mean = ss[0] / cnt;
        float var  = sq[0] / cnt - mean * mean;
        stats[n * 2 + 0] = mean;
        stats[n * 2 + 1] = var;
    }
}

__global__ void __launch_bounds__(256)
bn_apply_bf16(const float* __restrict__ H, const float* __restrict__ stats,
              const float* __restrict__ g, const float* __restrict__ be,
              bf16_t* __restrict__ Y, int Ncols) {
    const int row = blockIdx.x;
    const int n = row % 49;
    const float mean = stats[n * 2 + 0];
    const float var  = stats[n * 2 + 1];
    const float sc = g[n] * rsqrtf(var + 1e-5f);
    const float sh = be[n] - sc * mean;
    const float* hp = H + (size_t)row * Ncols;
    bf16_t* yp = Y + (size_t)row * Ncols;
    for (int c = threadIdx.x; c < Ncols; c += 256)
        yp[c] = (bf16_t)(sc * hp[c] + sh);
}

__global__ void f32_to_bf16(const float* __restrict__ X, bf16_t* __restrict__ Y, size_t n) {
    size_t i = (size_t)blockIdx.x * blockDim.x + threadIdx.x;
    if (i < n) Y[i] = (bf16_t)X[i];
}

// ---------------------------------------------------------------------------
// scores[n,m] = sum_b sum_d f1[b,n,d]*f2[b,m,d].  49x49 padded to 64x64.
// One block of 16 waves; wave w owns 16x16 tile (w>>2, w&3). K = 128 * 512.
// ---------------------------------------------------------------------------
__global__ void __launch_bounds__(512)
scores_wmma(const bf16_t* __restrict__ F1, const bf16_t* __restrict__ F2,
            float* __restrict__ S, int B, int N, int D) {
    const int w    = threadIdx.x >> 5;
    const int lane = threadIdx.x & 31;
    const int m0 = (w >> 2) << 4;
    const int n0 = (w & 3) << 4;
    const int arow = m0 + (lane & 15);
    const int akb  = (lane >> 4) << 3;
    const int bcol = n0 + (lane & 15);
    const int bkb  = (lane >> 4) << 3;
    const bool av = arow < N;
    const bool bv = bcol < N;

    v8f acc = {};
    for (int b = 0; b < B; ++b) {
        const bf16_t* f1b = F1 + (size_t)b * N * D;
        const bf16_t* f2b = F2 + (size_t)b * N * D;
        for (int k0 = 0; k0 < D; k0 += 32) {
            v16bf a = {};
            v16bf bb = {};
            if (av) a  = frag_a(f1b + (size_t)arow * D + k0 + akb);
            if (bv) bb = frag_b(f2b + (size_t)bcol * D + k0 + bkb);
            acc = __builtin_amdgcn_wmma_f32_16x16x32_bf16(false, a, false, bb, (short)0, acc, false, false);
        }
    }
    const int rbase = m0 + ((lane >> 4) << 3);
    const int col   = n0 + (lane & 15);
#pragma unroll
    for (int r = 0; r < 8; ++r) S[(rbase + r) * 64 + col] = acc[r];
}

// M[n,m] = vmat[n,m] + tanh( sum_j S[n,j]*wlin[m,j] + blin[m] )
__global__ void tanh_lin(const float* __restrict__ S, const float* __restrict__ wlin,
                         const float* __restrict__ blin, const float* __restrict__ vmat,
                         float* __restrict__ Mout) {
    int idx = blockIdx.x * blockDim.x + threadIdx.x;
    if (idx >= 49 * 49) return;
    int n = idx / 49, m = idx % 49;
    float s = blin[m];
    for (int j = 0; j < 49; ++j) s += S[n * 64 + j] * wlin[m * 49 + j];
    Mout[idx] = vmat[idx] + tanhf(s);
}

// Row r<49: softmax over row r of M; r>=49: softmax over row (r-49) of M^T.
// Top-20 by repeated argmax (ties -> lowest index, matching lax.top_k).
__global__ void softmax_topk(const float* __restrict__ Mmat,
                             float* __restrict__ topv, int* __restrict__ topi) {
    if (threadIdx.x != 0) return;
    const int row = blockIdx.x;
    const bool tr = row >= 49;
    const int n = tr ? row - 49 : row;
    float vals[49];
    for (int m = 0; m < 49; ++m) vals[m] = tr ? Mmat[m * 49 + n] : Mmat[n * 49 + m];
    float mx = vals[0];
    for (int m = 1; m < 49; ++m) mx = vals[m] > mx ? vals[m] : mx;
    float sum = 0.f;
    for (int m = 0; m < 49; ++m) { vals[m] = expf(vals[m] - mx); sum += vals[m]; }
    const float inv = 1.f / sum;
    for (int m = 0; m < 49; ++m) vals[m] *= inv;
    for (int k = 0; k < 20; ++k) {
        int bi = 0; float bvv = vals[0];
        for (int m = 1; m < 49; ++m) if (vals[m] > bvv) { bvv = vals[m]; bi = m; }
        topv[row * 20 + k] = bvv;
        topi[row * 20 + k] = bi;
        vals[bi] = -1.f;
    }
}

// ---------------------------------------------------------------------------
// Fused top-k gather + residual + LayerNorm(ddof=1).
// One block per (b,n). out row index = b*98 + out_off + n.
// ---------------------------------------------------------------------------
__global__ void __launch_bounds__(256)
fusion_ln(const float* __restrict__ Xbase, const float* __restrict__ Gsrc,
          const float* __restrict__ topv, const int* __restrict__ topi,
          const float* __restrict__ ln_g, const float* __restrict__ ln_b,
          float* __restrict__ rawout, float* __restrict__ Out, int out_off) {
    __shared__ float vv[20];
    __shared__ int   ii[20];
    __shared__ float red[256], red2[256];
    const int tid = threadIdx.x;
    const int b = blockIdx.x / 49, n = blockIdx.x % 49;
    if (tid < 20) { vv[tid] = topv[n * 20 + tid]; ii[tid] = topi[n * 20 + tid]; }
    __syncthreads();

    float acc[8];
    const float* xr = Xbase + ((size_t)b * 49 + n) * 2048;
#pragma unroll
    for (int i = 0; i < 8; ++i) acc[i] = xr[tid + (i << 8)];
    for (int k = 0; k < 20; ++k) {
        const float* gr = Gsrc + ((size_t)b * 49 + ii[k]) * 2048;
        const float w = vv[k];
#pragma unroll
        for (int i = 0; i < 8; ++i) acc[i] += w * gr[tid + (i << 8)];
    }
    if (rawout) {
        float* rp = rawout + ((size_t)b * 49 + n) * 2048;
#pragma unroll
        for (int i = 0; i < 8; ++i) rp[tid + (i << 8)] = acc[i];
    }
    float s = 0.f, q = 0.f;
#pragma unroll
    for (int i = 0; i < 8; ++i) { s += acc[i]; q += acc[i] * acc[i]; }
    red[tid] = s; red2[tid] = q; __syncthreads();
    for (int st = 128; st > 0; st >>= 1) {
        if (tid < st) { red[tid] += red[tid + st]; red2[tid] += red2[tid + st]; }
        __syncthreads();
    }
    const float tS = red[0], tQ = red2[0];
    const float mean = tS / 2048.f;
    const float var  = (tQ - tS * tS / 2048.f) / 2047.f;   // ddof = 1
    const float den  = sqrtf(var) + 1e-6f;                 // std + eps
    float* op = Out + ((size_t)b * 98 + out_off + n) * 2048;
#pragma unroll
    for (int i = 0; i < 8; ++i) {
        const int d = tid + (i << 8);
        op[d] = ln_g[d] * (acc[i] - mean) / den + ln_b[d];
    }
}

// ---------------------------------------------------------------------------
extern "C" void kernel_launch(void* const* d_in, const int* in_sizes, int n_in,
                              void* d_out, int out_size, void* d_ws, size_t ws_size,
                              hipStream_t stream) {
    const size_t BN = 6272, D = 2048, H1 = 1024, H2 = 512;
    const float* src[2] = {(const float*)d_in[0], (const float*)d_in[1]};
    const float* wlin = (const float*)d_in[18];
    const float* blin = (const float*)d_in[19];
    const float* ln_g = (const float*)d_in[20];
    const float* ln_b = (const float*)d_in[21];
    const float* vmat = (const float*)d_in[22];

    size_t off = 0;
    auto take = [&](size_t bytes) { size_t o = off; off += (bytes + 255) & ~(size_t)255; return o; };
    const size_t o_x   = take(BN * D * 2);
    const size_t o_wa  = take(H1 * D * 2);
    const size_t o_wb  = take(H2 * H1 * 2);
    const size_t o_h1  = take(BN * H1 * 4);   // unioned with s1_raw (see below)
    const size_t o_h1n = take(BN * H1 * 2);
    const size_t o_h2  = take(BN * H2 * 4);
    const size_t o_f1  = take(BN * H2 * 2);
    const size_t o_f2  = take(BN * H2 * 2);
    const size_t o_st  = take(49 * 2 * 4);
    const size_t o_S   = take(64 * 64 * 4);
    const size_t o_M   = take(49 * 49 * 4);
    const size_t o_tv  = take(98 * 20 * 4);
    const size_t o_ti  = take(98 * 20 * 4);
    const size_t o_s1  = o_h1;                // h1+h1n+h2 region == BN*D*4 exactly
    if (ws_size < off) return;

    char* ws = (char*)d_ws;
    bf16_t* x_bf   = (bf16_t*)(ws + o_x);
    bf16_t* wa_bf  = (bf16_t*)(ws + o_wa);
    bf16_t* wb_bf  = (bf16_t*)(ws + o_wb);
    float*  h1     = (float*)(ws + o_h1);
    bf16_t* h1n    = (bf16_t*)(ws + o_h1n);
    float*  h2     = (float*)(ws + o_h2);
    bf16_t* f_bf[2]= {(bf16_t*)(ws + o_f1), (bf16_t*)(ws + o_f2)};
    float*  stats  = (float*)(ws + o_st);
    float*  Sbuf   = (float*)(ws + o_S);
    float*  Mmat   = (float*)(ws + o_M);
    float*  topv   = (float*)(ws + o_tv);
    int*    topi   = (int*)(ws + o_ti);
    float*  s1raw  = (float*)(ws + o_s1);
    float*  out    = (float*)d_out;

    auto cdiv = [](size_t a, size_t b) { return (a + b - 1) / b; };

    for (int p = 0; p < 2; ++p) {
        const int base = 2 + p * 8;
        const float* Wa  = (const float*)d_in[base + 0];
        const float* ba  = (const float*)d_in[base + 1];
        const float* ga  = (const float*)d_in[base + 2];
        const float* bea = (const float*)d_in[base + 3];
        const float* Wb  = (const float*)d_in[base + 4];
        const float* bb  = (const float*)d_in[base + 5];
        const float* gb  = (const float*)d_in[base + 6];
        const float* beb = (const float*)d_in[base + 7];

        f32_to_bf16<<<cdiv(BN * D, 256), 256, 0, stream>>>(src[p], x_bf, BN * D);
        f32_to_bf16<<<cdiv(H1 * D, 256), 256, 0, stream>>>(Wa, wa_bf, H1 * D);
        f32_to_bf16<<<cdiv(H2 * H1, 256), 256, 0, stream>>>(Wb, wb_bf, H2 * H1);

        // 128x64 tiles: grid = 49 * (Ncols/64)
        gemm_bias_relu_bf16<<<49 * (int)(H1 / 64), 256, 0, stream>>>(x_bf, wa_bf, ba, h1,
                                                                     (int)H1, (int)D);
        bn_reduce<<<49, 256, 0, stream>>>(h1, stats, (int)H1);
        bn_apply_bf16<<<(int)BN, 256, 0, stream>>>(h1, stats, ga, bea, h1n, (int)H1);

        gemm_bias_relu_bf16<<<49 * (int)(H2 / 64), 256, 0, stream>>>(h1n, wb_bf, bb, h2,
                                                                     (int)H2, (int)H1);
        bn_reduce<<<49, 256, 0, stream>>>(h2, stats, (int)H2);
        bn_apply_bf16<<<(int)BN, 256, 0, stream>>>(h2, stats, gb, beb, f_bf[p], (int)H2);
    }

    scores_wmma<<<1, 512, 0, stream>>>(f_bf[0], f_bf[1], Sbuf, 128, 49, (int)H2);
    tanh_lin<<<(int)cdiv(49 * 49, 256), 256, 0, stream>>>(Sbuf, wlin, blin, vmat, Mmat);
    softmax_topk<<<98, 32, 0, stream>>>(Mmat, topv, topi);

    // s1 = fuse(src1 <- gather src2 via top-20 of softmax(M)); writes raw + LN half 0
    fusion_ln<<<(int)BN, 256, 0, stream>>>(src[0], src[1], topv, topi,
                                           ln_g, ln_b, s1raw, out, 0);
    // s2 = fuse(src2 <- gather s1raw via top-20 of softmax(M^T)); LN half 1
    fusion_ln<<<(int)BN, 256, 0, stream>>>(src[1], s1raw, topv + 49 * 20, topi + 49 * 20,
                                           ln_g, ln_b, nullptr, out, 49);
}
